// MultiHeadAttention_39359080300736
// MI455X (gfx1250) — matrix operand order repaired and checked by reading, not compiled, not run on hardware
//
#include <hip/hip_runtime.h>
#include <hip/hip_bf16.h>
#include <stdint.h>

#define B_  2
#define S_  2048
#define D_  1024
#define H_  16
#define DK_ 64
#define BS_ (B_ * S_)   // 4096
#define KB_ 32          // keys per attention inner step

typedef __attribute__((ext_vector_type(16))) _Float16 v16h;
typedef __attribute__((ext_vector_type(8)))  _Float16 v8h;
typedef __attribute__((ext_vector_type(8)))  float    v8f;

static __device__ __forceinline__ v16h combine8(v8h lo, v8h hi) {
    return __builtin_shufflevector(lo, hi, 0, 1, 2, 3, 4, 5, 6, 7,
                                           8, 9, 10, 11, 12, 13, 14, 15);
}

static __device__ __forceinline__ v8f wmma_f16(v16h a, v16h b, v8f c) {
    return __builtin_amdgcn_wmma_f32_16x16x32_f16(false, a, false, b,
                                                  (short)0, c, false, false);
}

// CDNA5 async copy: 16B global -> LDS, tracked by ASYNCcnt (ISA 08_async_tensor).
// vdst = VGPR with LDS byte address (generic shared addr low 32 bits, per ISA
// aperture rule), vaddr = 64-bit flat VA.
static __device__ __forceinline__ void async_copy_b128(_Float16* lds_dst,
                                                       const _Float16* g_src) {
    uint32_t loff = (uint32_t)(uintptr_t)lds_dst;
    uint64_t gaddr = (uint64_t)(uintptr_t)g_src;
    asm volatile("global_load_async_to_lds_b128 %0, %1, off"
                 :: "v"(loff), "v"(gaddr) : "memory");
}

static __device__ __forceinline__ void wait_asynccnt0() {
    asm volatile("s_wait_asynccnt 0x0" ::: "memory");
}

// ---------------------------------------------------------------------------
// f32 -> f16 elementwise conversion
// ---------------------------------------------------------------------------
__global__ void cvt_f32_to_f16(const float* __restrict__ in,
                               _Float16* __restrict__ out, int n) {
    int i = blockIdx.x * blockDim.x + threadIdx.x;
    if (i < n) out[i] = (_Float16)in[i];
}

// ---------------------------------------------------------------------------
// Y = X @ W^T + bias.  X:[4096,1024] f16, W:[1024,1024] f16 (row n = out feat).
// mode 0: f16 out, head-major  [B,H,S,DK]   (Q, K)
// mode 1: f16 out, transposed  [B,H,DK,S]   (V)
// mode 2: f32 out, plain       [M,N]        (final projection)
// ---------------------------------------------------------------------------
__global__ __launch_bounds__(256) void proj_gemm(const _Float16* __restrict__ X,
                                                 const _Float16* __restrict__ W,
                                                 const float* __restrict__ bias,
                                                 void* __restrict__ out, int mode) {
    const int wave = threadIdx.x >> 5;
    const int lane = threadIdx.x & 31;
    const int wm = wave & 3, wn = wave >> 2;
    const int m0 = blockIdx.x * 64 + wm * 16;
    const int n0 = blockIdx.y * 32 + wn * 16;
    const int lr = lane & 15;
    const int lh = lane >> 4;

    const _Float16* xrow = X + (size_t)(m0 + lr) * D_;
    const _Float16* wrow = W + (size_t)(n0 + lr) * D_;

    v8f c = {};
    for (int k0 = 0; k0 < D_; k0 += 32) {
        v8h  alo = *(const v8h*)(xrow + k0 + lh * 8);
        v8h  ahi = *(const v8h*)(xrow + k0 + lh * 8 + 16);
        v16h a   = combine8(alo, ahi);
        v16h b   = combine8(*(const v8h*)(wrow + k0 + lh * 8), *(const v8h*)(wrow + k0 + lh * 8 + 16));
        c = wmma_f16(a, b, c);
    }

    const float bv = bias[n0 + lr];
#pragma unroll
    for (int j = 0; j < 8; j++) c[j] += bv;

    if (mode == 2) {
        float* O = (float*)out;
#pragma unroll
        for (int j = 0; j < 8; j++) {
            int m = m0 + j + lh * 8;
            O[(size_t)m * D_ + n0 + lr] = c[j];
        }
    } else {
        _Float16* O = (_Float16*)out;
        const int n  = n0 + lr;
        const int h  = n >> 6;
        const int dk = n & 63;
#pragma unroll
        for (int j = 0; j < 8; j++) {
            int m  = m0 + j + lh * 8;
            int bb = m >> 11;
            int s  = m & 2047;
            size_t idx;
            if (mode == 0)
                idx = (((size_t)(bb * H_ + h)) * S_ + s) * DK_ + dk;
            else
                idx = (((size_t)(bb * H_ + h)) * DK_ + dk) * S_ + s;
            O[idx] = (_Float16)c[j];
        }
    }
}

// ---------------------------------------------------------------------------
// Flash attention. Grid: B*H*(S/64) blocks, 128 threads = 4 waves on the SAME
// (b,h); K/V tiles are staged once per block per 32-key step into LDS with
// async global->LDS copies (double-buffered), shared by all 4 waves.
// Qh,Kh: [B,H,S,DK] f16. Vt: [B,H,DK,S] f16. ctx: [B,S,D] f16.
// ---------------------------------------------------------------------------
__global__ __launch_bounds__(128) void flash_attn(const _Float16* __restrict__ Qh,
                                                  const _Float16* __restrict__ Kh,
                                                  const _Float16* __restrict__ Vt,
                                                  const int* __restrict__ mask,
                                                  _Float16* __restrict__ ctx) {
    __shared__ __align__(16) _Float16 kbuf[2][KB_ * DK_];  // [key][dk], 4KB each
    __shared__ __align__(16) _Float16 vbuf[2][DK_ * KB_];  // [dk][key], 4KB each
    __shared__ __align__(16) float    sc_lds[4][16 * 32];
    __shared__ __align__(16) _Float16 p_lds[4][16 * 32];
    __shared__ float row_scale[4][16];

    const int tid  = threadIdx.x;
    const int wave = tid >> 5;
    const int lane = tid & 31;
    const int bh     = blockIdx.x >> 5;
    const int qblock = blockIdx.x & 31;
    const int b = bh >> 4;
    const int h = bh & 15;
    const int qbase = qblock * 64 + wave * 16;

    const _Float16* Qp = Qh + ((size_t)bh * S_ + qbase) * DK_;
    const _Float16* Kp = Kh + (size_t)bh * S_ * DK_;
    const _Float16* Vp = Vt + (size_t)bh * DK_ * S_;
    const int*      Mp = mask + (size_t)b * S_ * S_;

    const int lr = lane & 15;
    const int lh = lane >> 4;

    // Q A-fragments; softmax scale 1/sqrt(64)=0.125 folded in (exact in f16)
    v16h qa[2];
    {
        const _Float16* rowp = Qp + (size_t)lr * DK_;
#pragma unroll
        for (int cc = 0; cc < 2; cc++) {
            v8h lo = *(const v8h*)(rowp + cc * 32 + lh * 8);
            v8h hi = *(const v8h*)(rowp + cc * 32 + lh * 8 + 16);
            v16h a = combine8(lo, hi);
#pragma unroll
            for (int i = 0; i < 16; i++) a[i] = a[i] * (_Float16)0.125f;
            qa[cc] = a;
        }
    }

    // Stage one 32-key step: K tile 4KB contiguous (256 x 16B chunks),
    // V tile 64 rows x 64B from the pre-transposed [dk][S] layout.
    auto stage = [&](int bufi, int kt) {
        const _Float16* kg = Kp + (size_t)kt * DK_;
        _Float16* kl = &kbuf[bufi][0];
        async_copy_b128(kl + (size_t)tid * 8,          kg + (size_t)tid * 8);
        async_copy_b128(kl + (size_t)(tid + 128) * 8,  kg + (size_t)(tid + 128) * 8);
        const _Float16* vg = Vp + kt;
        _Float16* vl = &vbuf[bufi][0];
        const int i0 = tid, i1 = tid + 128;
        async_copy_b128(vl + (i0 >> 2) * KB_ + (i0 & 3) * 8,
                        vg + (size_t)(i0 >> 2) * S_ + (i0 & 3) * 8);
        async_copy_b128(vl + (i1 >> 2) * KB_ + (i1 & 3) * 8,
                        vg + (size_t)(i1 >> 2) * S_ + (i1 & 3) * 8);
    };

    v8f o0 = {}, o1 = {}, o2 = {}, o3 = {};
    float mrow = -3.0e38f, lsum = 0.0f;   // lanes 0..15: stats for row qbase+lane

    stage(0, 0);
    int bufi = 0;
    for (int kt = 0; kt < S_; kt += KB_) {
        wait_asynccnt0();     // this wave's staged copies (issued last step) done
        __syncthreads();      // ... and every other wave's too
        if (kt + KB_ < S_) stage(bufi ^ 1, kt + KB_);

        // ---- scores: two 16x16 tiles, each a chain of two k=32 WMMAs ----
#pragma unroll
        for (int t = 0; t < 2; t++) {
            v8f c = {};
#pragma unroll
            for (int cc = 0; cc < 2; cc++) {
                const _Float16* kp = &kbuf[bufi][(t * 16 + lr) * DK_ + cc * 32 + lh * 8];
                v16h bfrag = combine8(*(const v8h*)kp, *(const v8h*)(kp + 16));
                c = wmma_f16(qa[cc], bfrag, c);
            }
#pragma unroll
            for (int j = 0; j < 8; j++)
                sc_lds[wave][(j + lh * 8) * 32 + t * 16 + lr] = c[j];
        }
        __syncthreads();

        // ---- streaming softmax: lane r owns query row qbase+r ----
        if (lane < 16) {
            const int q = qbase + lane;
            float tmax = -3.0e38f;
#pragma unroll
            for (int i = 0; i < 32; i++) {
                float s = sc_lds[wave][lane * 32 + i];
                if (Mp[(size_t)q * S_ + kt + i] == 0) s = -1.0e9f;
                sc_lds[wave][lane * 32 + i] = s;
                tmax = fmaxf(tmax, s);
            }
            const float nm   = fmaxf(mrow, tmax);
            const float corr = __expf(mrow - nm);
            float ssum = 0.0f;
#pragma unroll
            for (int i = 0; i < 32; i++) {
                float p = __expf(sc_lds[wave][lane * 32 + i] - nm);
                ssum += p;
                p_lds[wave][lane * 32 + i] = (_Float16)p;
            }
            lsum = lsum * corr + ssum;
            mrow = nm;
            row_scale[wave][lane] = corr;
        }
        __syncthreads();

        // ---- rescale accumulators ----
#pragma unroll
        for (int j = 0; j < 8; j++) {
            const float sc = row_scale[wave][j + lh * 8];
            o0[j] *= sc; o1[j] *= sc; o2[j] *= sc; o3[j] *= sc;
        }

        // ---- P A-fragment from LDS ----
        v16h pa;
        {
            const _Float16* pp = &p_lds[wave][lr * 32 + lh * 8];
            pa = combine8(*(const v8h*)pp, *(const v8h*)(pp + 16));
        }

        // ---- O += P @ V over 4 dk-tiles (B-fragments from staged LDS) ----
        {
            v16h v0t = combine8(*(const v8h*)&vbuf[bufi][(0 * 16 + lr) * KB_ + lh * 8], *(const v8h*)&vbuf[bufi][(0 * 16 + lr) * KB_ + lh * 8 + 16]);
            v16h v1t = combine8(*(const v8h*)&vbuf[bufi][(1 * 16 + lr) * KB_ + lh * 8], *(const v8h*)&vbuf[bufi][(1 * 16 + lr) * KB_ + lh * 8 + 16]);
            v16h v2t = combine8(*(const v8h*)&vbuf[bufi][(2 * 16 + lr) * KB_ + lh * 8], *(const v8h*)&vbuf[bufi][(2 * 16 + lr) * KB_ + lh * 8 + 16]);
            v16h v3t = combine8(*(const v8h*)&vbuf[bufi][(3 * 16 + lr) * KB_ + lh * 8], *(const v8h*)&vbuf[bufi][(3 * 16 + lr) * KB_ + lh * 8 + 16]);
            o0 = wmma_f16(pa, v0t, o0);
            o1 = wmma_f16(pa, v1t, o1);
            o2 = wmma_f16(pa, v2t, o2);
            o3 = wmma_f16(pa, v3t, o3);
        }
        bufi ^= 1;
    }

    // ---- finalize: divide by row sums, scatter to [B,S,D] f16 ----
    if (lane < 16) row_scale[wave][lane] = lsum;
    __syncthreads();

#pragma unroll
    for (int j = 0; j < 8; j++) {
        const float inv = 1.0f / row_scale[wave][j + lh * 8];
        const int m = qbase + j + lh * 8;
        _Float16* outp = ctx + ((size_t)(b * S_ + m)) * D_ + h * DK_;
        outp[0 * 16 + lr] = (_Float16)(o0[j] * inv);
        outp[1 * 16 + lr] = (_Float16)(o1[j] * inv);
        outp[2 * 16 + lr] = (_Float16)(o2[j] * inv);
        outp[3 * 16 + lr] = (_Float16)(o3[j] * inv);
    }
}

// ---------------------------------------------------------------------------
extern "C" void kernel_launch(void* const* d_in, const int* in_sizes, int n_in,
                              void* d_out, int out_size, void* d_ws, size_t ws_size,
                              hipStream_t stream) {
    const float* query = (const float*)d_in[0];
    const float* key   = (const float*)d_in[1];
    const float* value = (const float*)d_in[2];
    const int*   mask  = (const int*)d_in[3];
    const float* Wq = (const float*)d_in[4];
    const float* bq = (const float*)d_in[5];
    const float* Wk = (const float*)d_in[6];
    const float* bk = (const float*)d_in[7];
    const float* Wv = (const float*)d_in[8];
    const float* bv = (const float*)d_in[9];
    const float* Wo = (const float*)d_in[10];
    const float* bo = (const float*)d_in[11];
    float* out = (float*)d_out;

    char* ws = (char*)d_ws;
    size_t off = 0;
    auto take = [&](size_t bytes) -> char* {
        char* p = ws + off;
        off += (bytes + 255) & ~(size_t)255;
        return p;
    };
    const size_t XN = (size_t)BS_ * D_;
    const size_t WN = (size_t)D_ * D_;

    _Float16* q16  = (_Float16*)take(XN * 2);
    _Float16* k16  = (_Float16*)take(XN * 2);
    _Float16* v16  = (_Float16*)take(XN * 2);
    _Float16* wq16 = (_Float16*)take(WN * 2);
    _Float16* wk16 = (_Float16*)take(WN * 2);
    _Float16* wv16 = (_Float16*)take(WN * 2);
    _Float16* wo16 = (_Float16*)take(WN * 2);
    _Float16* Qh   = (_Float16*)take(XN * 2);   // [B,H,S,DK]
    _Float16* Kh   = (_Float16*)take(XN * 2);   // [B,H,S,DK]
    _Float16* Vt   = (_Float16*)take(XN * 2);   // [B,H,DK,S]
    _Float16* ctx  = (_Float16*)take(XN * 2);   // [B,S,D]

    const int cvtBlk = 256;
    cvt_f32_to_f16<<<(int)((XN + cvtBlk - 1) / cvtBlk), cvtBlk, 0, stream>>>(query, q16, (int)XN);
    cvt_f32_to_f16<<<(int)((XN + cvtBlk - 1) / cvtBlk), cvtBlk, 0, stream>>>(key,   k16, (int)XN);
    cvt_f32_to_f16<<<(int)((XN + cvtBlk - 1) / cvtBlk), cvtBlk, 0, stream>>>(value, v16, (int)XN);
    cvt_f32_to_f16<<<(int)((WN + cvtBlk - 1) / cvtBlk), cvtBlk, 0, stream>>>(Wq, wq16, (int)WN);
    cvt_f32_to_f16<<<(int)((WN + cvtBlk - 1) / cvtBlk), cvtBlk, 0, stream>>>(Wk, wk16, (int)WN);
    cvt_f32_to_f16<<<(int)((WN + cvtBlk - 1) / cvtBlk), cvtBlk, 0, stream>>>(Wv, wv16, (int)WN);
    cvt_f32_to_f16<<<(int)((WN + cvtBlk - 1) / cvtBlk), cvtBlk, 0, stream>>>(Wo, wo16, (int)WN);

    dim3 pgrid(BS_ / 64, D_ / 32);
    proj_gemm<<<pgrid, 256, 0, stream>>>(q16, wq16, bq, (void*)Qh, 0);
    proj_gemm<<<pgrid, 256, 0, stream>>>(k16, wk16, bk, (void*)Kh, 0);
    proj_gemm<<<pgrid, 256, 0, stream>>>(v16, wv16, bv, (void*)Vt, 1);

    flash_attn<<<B_ * H_ * (S_ / 64), 128, 0, stream>>>(Qh, Kh, Vt, mask, ctx);

    proj_gemm<<<pgrid, 256, 0, stream>>>(ctx, wo16, bo, (void*)out, 2);
}
